// HD95Loss_56779467653752
// MI455X (gfx1250) — compile-verified, matrix operand order checked by reading, not run on hardware
//
#include <hip/hip_runtime.h>
#include <hip/hip_bf16.h>

// ---------------------------------------------------------------------------
// HD95 on gfx1250 (MI455X, wave32).
//
// Pipeline (all deterministic, all on `stream`):
//   1) init:     zero coord pads / histogram / counters, 0xFFFFFFFF min arrays
//   2) compact:  ordered (row-major) compaction of >0.5 pixels, coords as
//                packed f16 pairs + counts (1024-thread LDS scan)
//   3) dist:     per (i-tile, i-block, j-block, img): V_WMMA_F32_16X16X32_F16
//                computes 16x16 dot-product tiles; d^2 = |p|^2+|t|^2-2*dot is
//                an EXACT integer (coords<=95 exact in f16; products exact in
//                f32) -> kept as uint32. DEPTH-2 software pipeline: the WMMA
//                issued in iteration n is consumed in iteration n+2, so its
//                4-co-exec-slot RAW hazard window (ISA 7.12.1) is filled with
//                a full iteration of independent VALU instead of v_nops.
//                Row-mins reduced in-regs via shfl_xor + LDS across 4 waves
//                (direct store); col-mins via global atomicMin(uint).
//                Sentinel 0xFFFFFFFF == "inf/dropped".
//   4) hist:     histogram of finite block-min d^2 values (18051 bins) + count
//   5) final:    exact 95th percentile by histogram prefix scan + linear
//                interpolation (matches nanquantile), mean over valid pairs.
// ---------------------------------------------------------------------------

#define IMG_W 96
#define NPIX  (96 * 96)        // 9216
#define BQ    1000             // reference cdist block size
#define K10   10000            // padded point capacity (ceil(9216/1000)*1000)
#define NB    10               // K10 / BQ
#define KPAD  10016            // coord array alloc (max index 10007)
#define NBINS 18051            // d^2 in [0, 95^2+95^2] = [0, 18050]
#define NBINS_AL 18052

// per-image uint offsets inside workspace
#define OFS_PC   0                         // pred coords (packed 2xf16), KPAD
#define OFS_TC   (KPAD)                    // true coords, KPAD
#define OFS_RM   (2 * KPAD)                // row mins  [K10][NB]
#define OFS_CM   (2 * KPAD + K10 * NB)     // col mins  [NB][K10]
#define OFS_HIST (2 * KPAD + 2 * K10 * NB) // histogram [NBINS_AL]
#define OFS_CNT  (OFS_HIST + NBINS_AL)     // [0]=cnt_p [1]=cnt_t [2]=n_finite
#define PER_IMG  (OFS_CNT + 4)             // ~238088 uints (~0.95 MB) / image

typedef __attribute__((ext_vector_type(16))) _Float16 v16h;
typedef __attribute__((ext_vector_type(8)))  float    v8f;

__device__ __forceinline__ _Float16 ush_as_h(unsigned short u) {
    union { unsigned short u; _Float16 h; } x; x.u = u; return x.h;
}
__device__ __forceinline__ unsigned short h_as_ush(_Float16 h) {
    union { unsigned short u; _Float16 h; } x; x.h = h; return x.u;
}

// ---------------------------------------------------------------------------
// 1) workspace init
// ---------------------------------------------------------------------------
__global__ void hd95_init_kernel(unsigned* ws) {
    int i = blockIdx.x * blockDim.x + threadIdx.x;
    if (i >= 2 * PER_IMG) return;
    int ofs = i % PER_IMG;
    unsigned v = 0u;
    if (ofs >= OFS_RM && ofs < OFS_HIST) v = 0xFFFFFFFFu;  // "inf" sentinel
    ws[i] = v;
}

// ---------------------------------------------------------------------------
// 2) ordered compaction (row-major == torch.nonzero order)
//    blockIdx.x in [0,4): img = b>>1, role = b&1 (0=pred/input, 1=true/target)
// ---------------------------------------------------------------------------
__global__ __launch_bounds__(1024) void hd95_compact_kernel(
        const float* __restrict__ input, const float* __restrict__ target,
        unsigned* __restrict__ ws) {
    const int img  = blockIdx.x >> 1;
    const int role = blockIdx.x & 1;
    const float* src = (role ? target : input) + img * NPIX;
    unsigned* base = ws + img * PER_IMG + (role ? OFS_TC : OFS_PC);
    unsigned* cnt  = ws + img * PER_IMG + OFS_CNT + role;

    __shared__ unsigned ssum[1024];
    const int t  = threadIdx.x;
    const int p0 = t * 9;                       // 1024 * 9 == 9216 exactly
    unsigned m = 0, bits = 0;
#pragma unroll
    for (int k = 0; k < 9; ++k) {
        bool b = src[p0 + k] > 0.5f;
        bits |= b ? (1u << k) : 0u;
        m += b ? 1u : 0u;
    }
    ssum[t] = m;
    __syncthreads();
    // Hillis-Steele inclusive scan over 1024 per-thread counts
    for (int off = 1; off < 1024; off <<= 1) {
        unsigned v = (t >= off) ? ssum[t - off] : 0u;
        __syncthreads();
        ssum[t] += v;
        __syncthreads();
    }
    unsigned pos = ssum[t] - m;                 // exclusive prefix
#pragma unroll
    for (int k = 0; k < 9; ++k) {
        if (bits & (1u << k)) {
            int p = p0 + k;
            int r = p / IMG_W, c = p % IMG_W;   // pc = (idx//W, idx%W)
            unsigned short hr = h_as_ush((_Float16)(float)r);  // exact, <=95
            unsigned short hc = h_as_ush((_Float16)(float)c);
            base[pos++] = ((unsigned)hc << 16) | (unsigned)hr;
        }
    }
    if (t == 1023) *cnt = ssum[1023];
}

// ---------------------------------------------------------------------------
// 3) WMMA distance tiles + block mins
//    grid: x = 63 i-tiles (16 rows) inside an i-block, y = ib*NB+jb, z = img
//    block: 128 threads = 4 waves; waves stride over the 63 j-tiles of block jb
//    (every wave owns >= 15 tiles, so the 2-deep pipeline prologue is valid)
// ---------------------------------------------------------------------------
__global__ __launch_bounds__(128) void hd95_dist_kernel(unsigned* __restrict__ ws) {
    const int img = blockIdx.z;
    const int ib  = blockIdx.y / NB;
    const int jb  = blockIdx.y % NB;
    const int it  = blockIdx.x;                 // 0..62
    unsigned* base = ws + img * PER_IMG;

    const int cnt_p = (int)base[OFS_CNT + 0];
    const int cnt_t = (int)base[OFS_CNT + 1];
    const int row0   = ib * BQ + it * 16;
    const int rowEnd = min(ib * BQ + BQ, cnt_p);
    const int colBase = jb * BQ;
    const int colEnd  = min(colBase + BQ, cnt_t);
    if (row0 >= rowEnd || colBase >= colEnd) return;  // block-uniform exit

    const int tid = threadIdx.x;
    const int wave = tid >> 5;
    const int lane = tid & 31;
    const int laneHalf = lane & 15;
    const int hi = lane >> 4;
    const bool lo16 = (lane < 16);

    // --- A operand (fixed per workgroup): 16x32 f16, only K0/K1 nonzero.
    // lanes 0-15 hold M=lane (K0,K1 in halfs 0,1); lanes 16-31 hold K=8.. -> 0
    unsigned pk = base[OFS_PC + row0 + laneHalf];
    _Float16 ax = ush_as_h((unsigned short)(pk & 0xFFFFu));
    _Float16 ay = ush_as_h((unsigned short)(pk >> 16));
    v16h A;
#pragma unroll
    for (int i = 0; i < 16; ++i) A[i] = (_Float16)0.0f;
    A[0] = lo16 ? ax : (_Float16)0.0f;
    A[1] = lo16 ? ay : (_Float16)0.0f;

    // |p|^2 and loop-invariant row-validity for this lane's 8 output rows
    float p2v[8];
    bool  rowValid[8];
#pragma unroll
    for (int v = 0; v < 8; ++v) {
        unsigned q = base[OFS_PC + row0 + v + 8 * hi];
        float fx = (float)ush_as_h((unsigned short)(q & 0xFFFFu));
        float fy = (float)ush_as_h((unsigned short)(q >> 16));
        p2v[v] = fx * fx + fy * fy;
        rowValid[v] = (row0 + v + 8 * hi) < rowEnd;
    }

    unsigned rmin[8];
#pragma unroll
    for (int v = 0; v < 8; ++v) rmin[v] = 0xFFFFFFFFu;
    unsigned* colm = base + OFS_CM + ib * K10;

    // consume one tile's D-matrix (d^2, mins) -- values exact ints in f32,
    // so a truncating u32 convert is bit-identical (no v_rndne needed)
    auto consume = [&](const v8f& C, float t2v, int colv) {
        const bool colValid = colv < colEnd;
        unsigned cmin = 0xFFFFFFFFu;
#pragma unroll
        for (int v = 0; v < 8; ++v) {
            float d2f = fmaxf(p2v[v] + t2v - 2.0f * C[v], 0.0f);
            unsigned d2u = (unsigned)d2f;       // exact: trunc == rn
            if (rowValid[v] && colValid) rmin[v] = min(rmin[v], d2u);
            if (rowValid[v]) cmin = min(cmin, d2u);
        }
        // column min over all 16 rows of the tile: combine lane pairs (xor 16)
        cmin = min(cmin, (unsigned)__shfl_xor((int)cmin, 16, 32));
        if (lo16 && colValid && cmin != 0xFFFFFFFFu)
            atomicMin(&colm[colv], cmin);       // combines across 63 i-tiles
    };

    // build B (32x16 f16: lanes 0-15 hold N=lane, K0/K1 in halfs 0,1) + t^2
    auto loadB = [&](int colv, v16h& B, float& t2v) {
        unsigned tk = base[OFS_TC + colv];      // max index 10007 < KPAD
        _Float16 bx = ush_as_h((unsigned short)(tk & 0xFFFFu));
        _Float16 by = ush_as_h((unsigned short)(tk >> 16));
        float tx = (float)bx, ty = (float)by;
        t2v = tx * tx + ty * ty;
#pragma unroll
        for (int i = 0; i < 16; ++i) B[i] = (_Float16)0.0f;
        B[0] = lo16 ? bx : (_Float16)0.0f;
        B[1] = lo16 ? by : (_Float16)0.0f;
    };

    const v8f CZ = {0.f, 0.f, 0.f, 0.f, 0.f, 0.f, 0.f, 0.f};

    // DEPTH-2 software pipeline: WMMA issued in iteration n is consumed in
    // iteration n+2 -> a full iteration of independent VALU sits inside the
    // WMMA_F16 -> VALU RAW hazard window (4 co-exec slots), no v_nops.
    int   colA = colBase + wave * 16 + laneHalf;
    float t2A;
    v8f   CA;
    {
        v16h B;
        loadB(colA, B, t2A);
        CA = __builtin_amdgcn_wmma_f32_16x16x32_f16(
                false, A, false, B, (short)0, CZ, false, false);
    }
    int   colB = colBase + (wave + 4) * 16 + laneHalf;
    float t2B;
    v8f   CB;
    {
        v16h B;
        loadB(colB, B, t2B);
        CB = __builtin_amdgcn_wmma_f32_16x16x32_f16(
                false, A, false, B, (short)0, CZ, false, false);
    }
    for (int jt = wave + 8; jt < 63; jt += 4) {
        int   colN = colBase + jt * 16 + laneHalf;
        float t2N;
        v16h  B;
        loadB(colN, B, t2N);
        v8f CN = __builtin_amdgcn_wmma_f32_16x16x32_f16(
                false, A, false, B, (short)0, CZ, false, false);
        consume(CA, t2A, colA);                 // issued two iterations ago
        CA = CB; t2A = t2B; colA = colB;
        CB = CN; t2B = t2N; colB = colN;
    }
    consume(CA, t2A, colA);                     // drain stage A
    consume(CB, t2B, colB);                     // drain stage B

    // row mins: reduce over 16 lanes of each half-wave (xor 1,2,4,8)
#pragma unroll
    for (int off = 1; off <= 8; off <<= 1) {
#pragma unroll
        for (int v = 0; v < 8; ++v)
            rmin[v] = min(rmin[v], (unsigned)__shfl_xor((int)rmin[v], off, 32));
    }
    __shared__ unsigned srm[4][16];
    if (laneHalf == 0) {
#pragma unroll
        for (int v = 0; v < 8; ++v) srm[wave][hi * 8 + v] = rmin[v];
    }
    __syncthreads();
    if (tid < 16) {
        unsigned m = min(min(srm[0][tid], srm[1][tid]),
                         min(srm[2][tid], srm[3][tid]));
        int r = row0 + tid;
        if (r < rowEnd) base[OFS_RM + r * NB + jb] = m;  // sole owner: store
    }
}

// ---------------------------------------------------------------------------
// 4) histogram of finite block-min d^2 values
// ---------------------------------------------------------------------------
__global__ void hd95_hist_kernel(unsigned* __restrict__ ws) {
    const int ENT = 2 * K10 * NB;               // row_mins + col_mins, contiguous
    int i = blockIdx.x * blockDim.x + threadIdx.x;
    if (i >= 2 * ENT) return;
    int img = i / ENT, e = i % ENT;
    unsigned* base = ws + img * PER_IMG;
    unsigned v = base[OFS_RM + e];
    if (v != 0xFFFFFFFFu) {
        if (v > 18050u) v = 18050u;
        atomicAdd(&base[OFS_HIST + v], 1u);
        atomicAdd(&base[OFS_CNT + 2], 1u);
    }
}

// ---------------------------------------------------------------------------
// 5) exact 95th percentile (linear interpolation) + mean over valid pairs
// ---------------------------------------------------------------------------
__global__ void hd95_final_kernel(unsigned* __restrict__ ws, float* __restrict__ out) {
    __shared__ float shd[2];
    __shared__ int   sok[2];
    int t = threadIdx.x;
    if (t < 2) {
        unsigned* base = ws + t * PER_IMG;
        unsigned cp = base[OFS_CNT + 0];
        unsigned ct = base[OFS_CNT + 1];
        unsigned n  = base[OFS_CNT + 2];
        int ok = (cp > 0u && ct > 0u && n > 0u) ? 1 : 0;
        float hd = 0.0f;
        if (ok) {
            double pos  = 0.95 * (double)(n - 1u);   // q*(n-1), linear method
            unsigned k0 = (unsigned)pos;
            double frac = pos - (double)k0;
            unsigned k1 = min(k0 + 1u, n - 1u);
            unsigned cum = 0, va = 18050u, vb = 18050u;
            bool fa = false;
            for (unsigned b = 0; b < NBINS; ++b) {   // rank selection scan
                cum += base[OFS_HIST + b];
                if (!fa && cum > k0) { va = b; fa = true; }
                if (cum > k1) { vb = b; break; }
            }
            float sva = sqrtf((float)va);
            float svb = sqrtf((float)vb);
            hd = sva + (float)frac * (svb - sva);
        }
        shd[t] = hd;
        sok[t] = ok;
    }
    __syncthreads();
    if (t == 0) {
        int nok = sok[0] + sok[1];
        float s = (sok[0] ? shd[0] : 0.0f) + (sok[1] ? shd[1] : 0.0f);
        out[0] = (nok > 0) ? (s / (float)nok) : __builtin_inff();
    }
}

// ---------------------------------------------------------------------------
extern "C" void kernel_launch(void* const* d_in, const int* in_sizes, int n_in,
                              void* d_out, int out_size, void* d_ws, size_t ws_size,
                              hipStream_t stream) {
    (void)in_sizes; (void)n_in; (void)out_size; (void)ws_size;
    const float* input  = (const float*)d_in[0];   // (2,96,96) f32
    const float* target = (const float*)d_in[1];   // (2,96,96) f32
    unsigned* ws  = (unsigned*)d_ws;               // needs 2*PER_IMG*4 ~ 1.9 MB
    float*    out = (float*)d_out;                 // scalar f32

    const int initN = 2 * PER_IMG;
    hd95_init_kernel<<<(initN + 255) / 256, 256, 0, stream>>>(ws);
    hd95_compact_kernel<<<4, 1024, 0, stream>>>(input, target, ws);
    hd95_dist_kernel<<<dim3(63, NB * NB, 2), dim3(128), 0, stream>>>(ws);
    const int histN = 2 * 2 * K10 * NB;
    hd95_hist_kernel<<<(histN + 255) / 256, 256, 0, stream>>>(ws);
    hd95_final_kernel<<<1, 32, 0, stream>>>(ws, out);
}
